// self_attn_64682207477773
// MI455X (gfx1250) — matrix operand order repaired and checked by reading, not compiled, not run on hardware
//
#include <hip/hip_runtime.h>
#include <hip/hip_bf16.h>

typedef __attribute__((ext_vector_type(16))) __bf16 v16bf;
typedef __attribute__((ext_vector_type(8)))  float  v8f;

#define NWIN  1024
#define SEQ   145
#define SPAD  160
#define NH    12
#define HD    64
#define DM    768
#define N3    2304
#define NDIST 532
#define BM    80
#define BN    128
#define KC    32

// Build a 16-element bf16 fragment from two 16B-aligned 8-half chunks.
__device__ __forceinline__ v16bf frag2(const __bf16* p0, const __bf16* p1) {
  v16bf r;
  *(float4*)&r          = *(const float4*)p0;
  *(((float4*)&r) + 1)  = *(const float4*)p1;
  return r;
}

// Async global->LDS 16-byte copy (CDNA5 GLOBAL_LOAD_ASYNC_TO_LDS_B128).
// LDS dest address = low 32 bits of the flat shared pointer (ISA 10.2).
__device__ __forceinline__ void async_cp16(void* lds, const void* gptr) {
  unsigned int l = (unsigned int)(unsigned long long)lds;
  asm volatile("global_load_async_to_lds_b128 %0, %1, off"
               :: "v"(l), "v"(gptr) : "memory");
}
__device__ __forceinline__ void async_wait0() {
  asm volatile("s_wait_asynccnt 0x0" ::: "memory");
}

__device__ __forceinline__ int relidx(int i, int j) {
  if (i == 0 && j == 0) return NDIST - 1;
  if (i == 0)           return NDIST - 3;
  if (j == 0)           return NDIST - 2;
  int yi = (i - 1) / 12, xi = (i - 1) % 12;
  int yj = (j - 1) / 12, xj = (j - 1) % 12;
  return (yi - yj + 11) * 23 + (xi - xj + 11);
}

// ---- Kernel 0: transpose+convert weights to bf16, N-major: Wt[n][k] ----
__global__ __launch_bounds__(256) void k_wconv(const float* __restrict__ Wq,
                                               const float* __restrict__ Wk,
                                               const float* __restrict__ Wv,
                                               __bf16* __restrict__ Wt) {
  int idx = blockIdx.x * 256 + threadIdx.x;
  if (idx >= N3 * DM) return;
  int n = idx / DM, k = idx - n * DM;
  const float* W = (n < DM) ? Wq : (n < 2 * DM ? Wk : Wv);
  int nn = (n < DM) ? n : (n < 2 * DM ? n - DM : n - 2 * DM);
  Wt[idx] = (__bf16)W[k * DM + nn];
}

// ---- Kernel 1: fused QKV projection GEMM (bf16 WMMA, fp32 accum) ----
// Y[m, n] = x[m, :] @ W[:, n]; n in [0,2304) selects Q/K/V slice.
// Q written pre-scaled by 0.125 (+bq); K plain; V written transposed (+bv).
__global__ __launch_bounds__(256) void k_qkv(const float* __restrict__ x,
                                             const float* __restrict__ bq,
                                             const float* __restrict__ bv,
                                             const __bf16* __restrict__ Wt,
                                             __bf16* __restrict__ Q,
                                             __bf16* __restrict__ K,
                                             __bf16* __restrict__ Vt) {
  __shared__ __bf16 xs[BM][KC + 8];   // 80 x 40 halves
  __shared__ __bf16 wsh[BN][KC + 8];  // 128 x 40 halves
  const int b   = blockIdx.z;
  const int m0  = blockIdx.y * BM;
  const int n0  = blockIdx.x * BN;
  const int tid = threadIdx.x;
  const int lane = tid & 31, wv = tid >> 5;
  const int nl = lane & 15, kg = lane >> 4;

  v8f acc[5] = {};

  for (int k0 = 0; k0 < DM; k0 += KC) {
    __syncthreads();   // previous tile fully consumed before overwriting
    // async-stage Wt tile: 128 rows(n) x 32(k) bf16, 16B chunks, no VGPR hop
#pragma unroll
    for (int it = 0; it < 2; ++it) {
      int i = tid + it * 256;
      int n = i >> 2, kk = (i & 3) * 8;
      async_cp16(&wsh[n][kk], Wt + (size_t)(n0 + n) * DM + k0 + kk);
    }
    // sync-stage x tile (fp32 -> bf16 conversion overlaps the async copy)
#pragma unroll
    for (int it = 0; it < 3; ++it) {
      int i = tid + it * 256;
      if (i < BM * (KC / 4)) {
        int m = i >> 3, kk = (i & 7) * 4;
        int s = m0 + m;
        float4 f = {0.f, 0.f, 0.f, 0.f};
        if (s < SEQ) {
          const float* px = x + ((size_t)b * SEQ + s) * DM + k0 + kk;
          f = *(const float4*)px;
          if (k0 + KC < DM) __builtin_prefetch(px + KC, 0, 3);
        }
        __bf16 t4[4] = {(__bf16)f.x, (__bf16)f.y, (__bf16)f.z, (__bf16)f.w};
        *(float2*)&xs[m][kk] = *(float2*)t4;
      }
    }
    async_wait0();
    __syncthreads();

    // B fragment for this wave's 16-column strip
    const __bf16* br = &wsh[wv * 16 + nl][kg * 8];
    v16bf bf = frag2(br, br + 16);
#pragma unroll
    for (int t = 0; t < 5; ++t) {
      const __bf16* pr = &xs[t * 16 + nl][kg * 8];
      v16bf af = frag2(pr, pr + 16);
      acc[t] = __builtin_amdgcn_wmma_f32_16x16x32_bf16(
          false, af, false, bf, (short)0, acc[t], false, false);
    }
  }

  // which output (Q/K/V) is uniform per block: BN=128 divides 768
  const int which = n0 / DM;
  const int nn    = (n0 - which * DM) + wv * 16 + nl;
  const int h     = nn >> 6;
  const int d     = nn & 63;
  const size_t rowbase = ((size_t)b * NH + h) * SPAD;
  if (which == 0) {
    const float bias = bq[nn];
#pragma unroll
    for (int t = 0; t < 5; ++t)
#pragma unroll
      for (int r = 0; r < 8; ++r) {
        int s = m0 + t * 16 + r + kg * 8;
        Q[(rowbase + s) * HD + d] = (__bf16)((acc[t][r] + bias) * 0.125f);
      }
  } else if (which == 1) {
#pragma unroll
    for (int t = 0; t < 5; ++t)
#pragma unroll
      for (int r = 0; r < 8; ++r) {
        int s = m0 + t * 16 + r + kg * 8;
        K[(rowbase + s) * HD + d] = (__bf16)acc[t][r];
      }
  } else {
    const float bias = bv[nn];
    const size_t vbase = (((size_t)b * NH + h) * HD + d) * SPAD;
#pragma unroll
    for (int t = 0; t < 5; ++t)
#pragma unroll
      for (int r = 0; r < 8; ++r) {
        int s = m0 + t * 16 + r + kg * 8;
        Vt[vbase + s] = (__bf16)(acc[t][r] + bias);
      }
  }
}

// ---- Kernel 2: attention per (window, head). 10 waves, 16 query rows each ----
__global__ __launch_bounds__(320) void k_attn(const __bf16* __restrict__ Q,
                                              const __bf16* __restrict__ K,
                                              const __bf16* __restrict__ Vt,
                                              const float* __restrict__ mask,
                                              const float* __restrict__ relTbl,
                                              float* __restrict__ out) {
  __shared__ __bf16 Ks[SPAD][HD + 8];          // keys row-major    (23.0 KB)
  __shared__ __bf16 Vs[HD][SPAD + 8];          // V transposed      (21.5 KB)
  __shared__ __bf16 Ps[10][16][SPAD + 8];      // per-wave probs    (53.8 KB)
  __shared__ float  bias_h[NDIST];

  const int h = blockIdx.x, b = blockIdx.y;
  const int tid = threadIdx.x;
  const int lane = tid & 31, wv = tid >> 5;
  const int nl = lane & 15, kg = lane >> 4;

  const size_t ho = (size_t)b * NH + h;
  const __bf16* Kg = K  + ho * SPAD * HD;
  const __bf16* Vg = Vt + ho * HD * SPAD;
  const __bf16* Qg = Q  + ho * SPAD * HD;

  // async-stage K and V^T straight into LDS (ASYNCcnt path)
#pragma unroll
  for (int it = 0; it < 4; ++it) {             // 160*8 / 320 = 4 exact
    int i = tid + it * 320;
    int m = i >> 3, c = (i & 7) * 8;
    async_cp16(&Ks[m][c], Kg + m * HD + c);
  }
#pragma unroll
  for (int it = 0; it < 4; ++it) {             // 64*20 / 320 = 4 exact
    int i = tid + it * 320;
    int dd = i / (SPAD / 8), c = (i % (SPAD / 8)) * 8;
    async_cp16(&Vs[dd][c], Vg + dd * SPAD + c);
  }
  for (int i = tid; i < NDIST; i += 320) bias_h[i] = relTbl[i * NH + h];
  async_wait0();
  __syncthreads();

  const int s0 = wv * 16;
  // Q fragments (Q pre-scaled by 1/sqrt(64) in kernel 1)
  v16bf aq[2];
  {
    const __bf16* qrow = Qg + (size_t)(s0 + nl) * HD;
#pragma unroll
    for (int c = 0; c < 2; ++c)
      aq[c] = frag2(qrow + c * 32 + kg * 8, qrow + c * 32 + kg * 8 + 16);
  }
  // scores: 16 x 160 via 10 key tiles
  float sc[10][8];
#pragma unroll
  for (int j = 0; j < 10; ++j) {
    v8f a = {};
#pragma unroll
    for (int c = 0; c < 2; ++c) {
      const __bf16* kr = &Ks[j * 16 + nl][c * 32 + kg * 8];
      v16bf bk = frag2(kr, kr + 16);
      a = __builtin_amdgcn_wmma_f32_16x16x32_bf16(
          false, aq[c], false, bk, (short)0, a, false, false);
    }
#pragma unroll
    for (int r = 0; r < 8; ++r) sc[j][r] = a[r];
  }
  // mask + relative bias + pad masking, then softmax per row
#pragma unroll
  for (int r = 0; r < 8; ++r) {
    int qi = s0 + r + kg * 8;
#pragma unroll
    for (int j = 0; j < 10; ++j) {
      int kj = j * 16 + nl;
      float v = sc[j][r];
      if (qi < SEQ) {
        if (kj < SEQ) {
          v += mask[((size_t)b * SEQ + qi) * SEQ + kj];
          v += bias_h[relidx(qi, kj)];
        } else v = -1e30f;
      } else v = 0.f;   // dead rows: keep finite
      sc[j][r] = v;
    }
    float mx = sc[0][r];
#pragma unroll
    for (int j = 1; j < 10; ++j) mx = fmaxf(mx, sc[j][r]);
#pragma unroll
    for (int o = 1; o < 16; o <<= 1) mx = fmaxf(mx, __shfl_xor(mx, o, 32));
    float sum = 0.f;
#pragma unroll
    for (int j = 0; j < 10; ++j) {
      float e = __expf(sc[j][r] - mx);
      sc[j][r] = e; sum += e;
    }
#pragma unroll
    for (int o = 1; o < 16; o <<= 1) sum += __shfl_xor(sum, o, 32);
    float inv = 1.0f / sum;
#pragma unroll
    for (int j = 0; j < 10; ++j) {
      float p = sc[j][r] * inv;
      Ps[wv][r + kg * 8][j * 16 + nl] = (__bf16)p;  // C-layout -> LDS
    }
  }
  asm volatile("s_wait_dscnt 0x0" ::: "memory");    // per-wave LDS RAW

  // context: P(16x160) @ V(160x64)
  v16bf ap[5];
#pragma unroll
  for (int c = 0; c < 5; ++c) {
    const __bf16* pr = &Ps[wv][nl][c * 32 + kg * 8];
    ap[c] = frag2(pr, pr + 16);
  }
#pragma unroll
  for (int t = 0; t < 4; ++t) {
    v8f a = {};
#pragma unroll
    for (int c = 0; c < 5; ++c) {
      const __bf16* vr = &Vs[t * 16 + nl][c * 32 + kg * 8];
      v16bf bv16 = frag2(vr, vr + 16);
      a = __builtin_amdgcn_wmma_f32_16x16x32_bf16(
          false, ap[c], false, bv16, (short)0, a, false, false);
    }
#pragma unroll
    for (int r = 0; r < 8; ++r) {
      int qi = s0 + r + kg * 8;
      if (qi < SEQ)
        out[((size_t)b * SEQ + qi) * DM + h * HD + t * 16 + nl] = a[r];
    }
  }
}

extern "C" void kernel_launch(void* const* d_in, const int* in_sizes, int n_in,
                              void* d_out, int out_size, void* d_ws, size_t ws_size,
                              hipStream_t stream) {
  const float* x    = (const float*)d_in[0];
  const float* mask = (const float*)d_in[1];
  const float* Wq   = (const float*)d_in[2];
  const float* bq   = (const float*)d_in[3];
  const float* Wk   = (const float*)d_in[4];
  const float* Wv   = (const float*)d_in[5];
  const float* bv   = (const float*)d_in[6];
  const float* rel  = (const float*)d_in[7];
  float* out = (float*)d_out;

  char* ws = (char*)d_ws;
  size_t o = 0;
  __bf16* Wt = (__bf16*)(ws + o); o += (size_t)N3 * DM * 2;  o = (o + 255) & ~(size_t)255;
  const size_t qkvBytes = (size_t)NWIN * NH * SPAD * HD * 2;
  __bf16* Qb = (__bf16*)(ws + o); o += qkvBytes;
  __bf16* Kb = (__bf16*)(ws + o); o += qkvBytes;
  __bf16* Vb = (__bf16*)(ws + o);

  k_wconv<<<(N3 * DM + 255) / 256, 256, 0, stream>>>(Wq, Wk, Wv, Wt);
  k_qkv<<<dim3(N3 / BN, SPAD / BM, NWIN), 256, 0, stream>>>(x, bq, bv, Wt, Qb, Kb, Vb);
  k_attn<<<dim3(NH, NWIN), 320, 0, stream>>>(Qb, Kb, Vb, mask, rel, out);
}
